// TimeSeriesGPT_45749991637173
// MI455X (gfx1250) — matrix operand order repaired and checked by reading, not compiled, not run on hardware
//
#include <hip/hip_runtime.h>
#include <hip/hip_bf16.h>

// =====================================================================
// TimeSeriesGPT on MI455X (gfx1250):
// Autoregressive decode rewritten as KV-cache incremental decoding.
// All D=512 / D_FF=2048 GEMMs use v_wmma_f32_16x16x32_bf16 with M=16
// (= batch) per wave-tile; weights converted once per launch to bf16
// in workspace (~25 MB -> resident in 192 MB L2 across the 31 steps).
// =====================================================================

typedef __bf16 bf16;
typedef __attribute__((ext_vector_type(8)))  __bf16 v8bf;
typedef __attribute__((ext_vector_type(16))) __bf16 v16bf;
typedef __attribute__((ext_vector_type(8)))  float  v8f;

#define Bsz   16
#define Ssz   32
#define DIN   8
#define Dm    512
#define DOUT  8
#define Lc    4
#define Hc    8
#define DHh   64
#define DFF   2048
#define NSTEPS 31

__device__ __forceinline__ bf16 f2bf(float f) {
  union { float f; unsigned u; } in; in.f = f;
  unsigned u = in.u + 0x7FFFu + ((in.u >> 16) & 1u); // round-to-nearest-even
  union { unsigned short s; bf16 b; } out; out.s = (unsigned short)(u >> 16);
  return out.b;
}

union ABfrag { v16bf v; v8bf p[2]; };

// ---------------------------------------------------------------------
// One wave computes a 16x16 f32 tile of A(16xK,bf16) @ W(KxN,bf16).
// A fragment (16-bit A 16x32 layout, ISA 7.12.2): lane m=lane&15,
// VGPRs 0-3 hold K = hi*8 + 0..7, VGPRs 4-7 hold K = 16 + hi*8 + 0..7.
// B fragment (16-bit B layout): lane <-> N column n0 + (lane&15),
// elements <-> K in the same order as A -> 16 strided (stride N) column reads per lane.
// ---------------------------------------------------------------------
__device__ __forceinline__ v8f wmma_tile(const bf16* __restrict__ A,
                                         const bf16* __restrict__ W,
                                         int K, int N, int n0, int lane)
{
  const int m  = lane & 15;
  const int hi = lane >> 4;
  v8f acc = {};
  const bf16* arow = A + m * K + hi * 8;
  const bf16* wrow = W + (hi * 8) * N + n0 + m;
  for (int kb = 0; kb < K; kb += 32) {
    ABfrag a;
    a.p[0] = *(const v8bf*)(arow + kb);
    a.p[1] = *(const v8bf*)(arow + kb + 16);
    v16bf b;
    for (int s = 0; s < 8; ++s) {
      b[s]     = wrow[(kb + s) * N];
      b[8 + s] = wrow[(kb + 16 + s) * N];
    }
    if (kb + 32 < K)                       // global_prefetch_b8: next K-block
      __builtin_prefetch(wrow + (kb + 32) * N, 0, 1);
    acc = __builtin_amdgcn_wmma_f32_16x16x32_bf16(false, a.v, false, b,
                                                  (short)0, acc, false, false);
  }
  return acc;
}

// ---------------------- f32 -> bf16 weight convert -------------------
__global__ void cvt_bf16_kernel(const float* __restrict__ src,
                                bf16* __restrict__ dst, int n)
{
  int i = blockIdx.x * blockDim.x + threadIdx.x;
  int stride = gridDim.x * blockDim.x;
  for (; i < n; i += stride) dst[i] = f2bf(src[i]);
}

// ---------------------- token embed + pos encoding -------------------
// h[b][d] = sum_i y[b][i]*Wt[i][d] + bt[d] + pe[t][d]
__global__ __launch_bounds__(256) void embed_kernel(
    const float* __restrict__ seq, const float* __restrict__ prev,
    const float* __restrict__ pe, const float* __restrict__ Wt,
    const float* __restrict__ bt,
    float* __restrict__ h, bf16* __restrict__ hb, int t)
{
  int idx = blockIdx.x * 256 + threadIdx.x;        // 0..8191
  int b = idx >> 9, d = idx & 511;
  float y[DIN];
  if (t == 0) {
    for (int i = 0; i < DIN; ++i) y[i] = seq[b * (Ssz * DIN) + i];
  } else {
    for (int i = 0; i < DIN; ++i) y[i] = prev[(b * NSTEPS + (t - 1)) * DOUT + i];
  }
  float acc = bt[d] + pe[t * Dm + d];
  for (int i = 0; i < DIN; ++i) acc += y[i] * Wt[i * Dm + d];
  h[idx]  = acc;
  hb[idx] = f2bf(acc);
}

// ---------------------- fused Q/K/V projection -----------------------
// 96 waves: wave = (mat in {Q,K,V}) x (32 N-tiles). K/V written into the
// per-head KV cache at position t: [(b*H+h)*S + t]*DH + dh.
__global__ __launch_bounds__(128) void gemm_qkv_kernel(
    const bf16* __restrict__ hb,
    const bf16* __restrict__ Wq, const bf16* __restrict__ Wk,
    const bf16* __restrict__ Wv,
    const float* __restrict__ bq, const float* __restrict__ bk,
    const float* __restrict__ bv,
    float* __restrict__ qout, float* __restrict__ kc, float* __restrict__ vc,
    int t)
{
  int lane = threadIdx.x & 31;
  int wid  = (blockIdx.x * blockDim.x + threadIdx.x) >> 5;   // 0..95
  int mat  = wid >> 5;                                       // 0=Q 1=K 2=V
  int n0   = (wid & 31) * 16;
  const bf16*  W    = (mat == 0) ? Wq : (mat == 1) ? Wk : Wv;
  const float* bias = (mat == 0) ? bq : (mat == 1) ? bk : bv;
  v8f acc = wmma_tile(hb, W, Dm, Dm, n0, lane);
  int hi = lane >> 4;
  int n  = n0 + (lane & 15);
  float bn = bias[n];
  int head = n >> 6, dh = n & 63;
  for (int r = 0; r < 8; ++r) {
    int m = r + 8 * hi;                 // C layout: VGPR r -> M = r + 8*hi
    float v = acc[r] + bn;
    if (mat == 0) {
      qout[m * Dm + n] = v;
    } else {
      float* dst = (mat == 1) ? kc : vc;
      dst[((m * Hc + head) * Ssz + t) * DHh + dh] = v;
    }
  }
}

// ---------------------- single-position causal attention -------------
// One wave per (b, head). Lane j owns key position j (j<=t active).
__global__ __launch_bounds__(32) void attn_kernel(
    const float* __restrict__ q, const float* __restrict__ kc,
    const float* __restrict__ vc, bf16* __restrict__ ctxb, int t)
{
  __shared__ float sp[32];
  int bid = blockIdx.x;
  int b = bid >> 3, hd = bid & 7;
  int j = threadIdx.x;
  const float* qv  = q  + b * Dm + hd * DHh;
  const float* Kbh = kc + (b * Hc + hd) * Ssz * DHh;
  const float* Vbh = vc + (b * Hc + hd) * Ssz * DHh;
  float s = -1e30f;
  if (j <= t) {
    float acc = 0.f;
    const float* Kj = Kbh + j * DHh;
    for (int dh = 0; dh < DHh; ++dh) acc += qv[dh] * Kj[dh];
    s = acc * 0.125f;                    // 1/sqrt(64)
  }
  sp[j] = s; __syncthreads();
  float mx = -1e30f;
  for (int jj = 0; jj <= t; ++jj) mx = fmaxf(mx, sp[jj]);
  __syncthreads();
  float e = (j <= t) ? expf(s - mx) : 0.f;
  sp[j] = e; __syncthreads();
  float sum = 0.f;
  for (int jj = 0; jj <= t; ++jj) sum += sp[jj];
  float inv = 1.f / sum;
  int dh0 = j * 2;                       // each lane produces 2 ctx dims
  float c0 = 0.f, c1 = 0.f;
  for (int jj = 0; jj <= t; ++jj) {
    float pw = sp[jj] * inv;
    c0 += pw * Vbh[jj * DHh + dh0];
    c1 += pw * Vbh[jj * DHh + dh0 + 1];
  }
  ctxb[b * Dm + hd * DHh + dh0]     = f2bf(c0);
  ctxb[b * Dm + hd * DHh + dh0 + 1] = f2bf(c1);
}

// ---------------------- generic WMMA GEMM epilogue -------------------
// out = A(16xK) @ W(KxN) + bias [+ resid] [relu], to f32 and/or bf16.
__global__ __launch_bounds__(128) void gemm_out_kernel(
    const bf16* __restrict__ A, const bf16* __restrict__ W,
    const float* __restrict__ bias, const float* __restrict__ resid,
    float* __restrict__ outf, bf16* __restrict__ outb,
    int K, int N, int relu)
{
  int lane = threadIdx.x & 31;
  int wid  = (blockIdx.x * blockDim.x + threadIdx.x) >> 5;
  int n0   = wid * 16;
  v8f acc = wmma_tile(A, W, K, N, n0, lane);
  int hi = lane >> 4;
  int n  = n0 + (lane & 15);
  float bn = bias[n];
  for (int r = 0; r < 8; ++r) {
    int m = r + 8 * hi;
    float v = acc[r] + bn;
    if (resid) v += resid[m * N + n];
    if (relu)  v = fmaxf(v, 0.f);
    if (outf) outf[m * N + n] = v;
    if (outb) outb[m * N + n] = f2bf(v);
  }
}

// ---------------------- layernorm over D=512 per row -----------------
__global__ __launch_bounds__(256) void layernorm_kernel(
    const float* __restrict__ x, const float* __restrict__ scale,
    const float* __restrict__ bias,
    float* __restrict__ h, bf16* __restrict__ hb)
{
  __shared__ float red[256];
  int b = blockIdx.x, tid = threadIdx.x;
  const float* row = x + b * Dm;
  float v0 = row[tid], v1 = row[tid + 256];
  red[tid] = v0 + v1; __syncthreads();
  for (int s = 128; s > 0; s >>= 1) {
    if (tid < s) red[tid] += red[tid + s];
    __syncthreads();
  }
  float mu = red[0] * (1.f / 512.f);
  __syncthreads();
  float d0 = v0 - mu, d1 = v1 - mu;
  red[tid] = d0 * d0 + d1 * d1; __syncthreads();
  for (int s = 128; s > 0; s >>= 1) {
    if (tid < s) red[tid] += red[tid + s];
    __syncthreads();
  }
  float rstd = rsqrtf(red[0] * (1.f / 512.f) + 1e-5f);
  float o0 = d0 * rstd * scale[tid]       + bias[tid];
  float o1 = d1 * rstd * scale[tid + 256] + bias[tid + 256];
  h[b * Dm + tid]        = o0;  h[b * Dm + tid + 256]  = o1;
  hb[b * Dm + tid]       = f2bf(o0);
  hb[b * Dm + tid + 256] = f2bf(o1);
}

// ---------------------- final 512 -> 8 projection --------------------
__global__ __launch_bounds__(128) void final_kernel(
    const float* __restrict__ h, const float* __restrict__ Wf,
    const float* __restrict__ bfv, float* __restrict__ out, int t)
{
  int b = threadIdx.x >> 3, o = threadIdx.x & 7;
  float acc = bfv[o];
  const float* hr = h + b * Dm;
  for (int d = 0; d < Dm; ++d) acc += hr[d] * Wf[d * DOUT + o];
  out[(b * NSTEPS + t) * DOUT + o] = acc;
}

// =====================================================================
extern "C" void kernel_launch(void* const* d_in, const int* in_sizes, int n_in,
                              void* d_out, int out_size, void* d_ws, size_t ws_size,
                              hipStream_t stream)
{
  const float* seq  = (const float*)d_in[0];
  const float* pe   = (const float*)d_in[1];
  const float* Wt   = (const float*)d_in[2];
  const float* bt   = (const float*)d_in[3];
  const float* Wq   = (const float*)d_in[4];
  const float* bq   = (const float*)d_in[5];
  const float* Wk   = (const float*)d_in[6];
  const float* bk   = (const float*)d_in[7];
  const float* Wv   = (const float*)d_in[8];
  const float* bv   = (const float*)d_in[9];
  const float* Wo   = (const float*)d_in[10];
  const float* bo   = (const float*)d_in[11];
  const float* ln1s = (const float*)d_in[12];
  const float* ln1b = (const float*)d_in[13];
  const float* W1   = (const float*)d_in[14];
  const float* b1   = (const float*)d_in[15];
  const float* W2   = (const float*)d_in[16];
  const float* b2   = (const float*)d_in[17];
  const float* ln2s = (const float*)d_in[18];
  const float* ln2b = (const float*)d_in[19];
  const float* Wf   = (const float*)d_in[20];
  const float* bfv  = (const float*)d_in[21];
  float* out = (float*)d_out;

  // ---- workspace layout (~34 MB) ----
  char* ws = (char*)d_ws;
  size_t off = 0;
  auto alloc = [&](size_t bytes) -> char* {
    char* p = ws + off;
    off = (off + bytes + 255) & ~(size_t)255;
    return p;
  };
  const size_t nDD  = (size_t)Lc * Dm * Dm;     // 1,048,576
  const size_t nDF  = (size_t)Lc * Dm * DFF;    // 4,194,304
  const size_t nKV  = (size_t)Lc * Bsz * Hc * Ssz * DHh; // 1,048,576 floats

  bf16*  Wq_b   = (bf16*)alloc(nDD * 2);
  bf16*  Wk_b   = (bf16*)alloc(nDD * 2);
  bf16*  Wv_b   = (bf16*)alloc(nDD * 2);
  bf16*  Wo_b   = (bf16*)alloc(nDD * 2);
  bf16*  W1_b   = (bf16*)alloc(nDF * 2);
  bf16*  W2_b   = (bf16*)alloc(nDF * 2);
  float* kcache = (float*)alloc(nKV * 4);
  float* vcache = (float*)alloc(nKV * 4);
  float* hbuf   = (float*)alloc((size_t)Bsz * Dm * 4);
  float* xbuf   = (float*)alloc((size_t)Bsz * Dm * 4);
  float* qbuf   = (float*)alloc((size_t)Bsz * Dm * 4);
  bf16*  hb     = (bf16*)alloc((size_t)Bsz * Dm * 2);
  bf16*  ctxb   = (bf16*)alloc((size_t)Bsz * Dm * 2);
  bf16*  f1b    = (bf16*)alloc((size_t)Bsz * DFF * 2);
  (void)ws_size; (void)in_sizes; (void)n_in; (void)out_size;

  // ---- convert weights to bf16 (per launch; deterministic) ----
  cvt_bf16_kernel<<<1024, 256, 0, stream>>>(Wq, Wq_b, (int)nDD);
  cvt_bf16_kernel<<<1024, 256, 0, stream>>>(Wk, Wk_b, (int)nDD);
  cvt_bf16_kernel<<<1024, 256, 0, stream>>>(Wv, Wv_b, (int)nDD);
  cvt_bf16_kernel<<<1024, 256, 0, stream>>>(Wo, Wo_b, (int)nDD);
  cvt_bf16_kernel<<<2048, 256, 0, stream>>>(W1, W1_b, (int)nDF);
  cvt_bf16_kernel<<<2048, 256, 0, stream>>>(W2, W2_b, (int)nDF);

  // ---- autoregressive decode: 31 steps with KV cache ----
  for (int t = 0; t < NSTEPS; ++t) {
    embed_kernel<<<32, 256, 0, stream>>>(seq, out, pe, Wt, bt, hbuf, hb, t);

    for (int l = 0; l < Lc; ++l) {
      const bf16* Wq_l = Wq_b + (size_t)l * Dm * Dm;
      const bf16* Wk_l = Wk_b + (size_t)l * Dm * Dm;
      const bf16* Wv_l = Wv_b + (size_t)l * Dm * Dm;
      const bf16* Wo_l = Wo_b + (size_t)l * Dm * Dm;
      const bf16* W1_l = W1_b + (size_t)l * Dm * DFF;
      const bf16* W2_l = W2_b + (size_t)l * DFF * Dm;
      float* kcL = kcache + (size_t)l * Bsz * Hc * Ssz * DHh;
      float* vcL = vcache + (size_t)l * Bsz * Hc * Ssz * DHh;

      // Q,K,V projections: 96 waves (3 mats x 32 N-tiles)
      gemm_qkv_kernel<<<24, 128, 0, stream>>>(
          hb, Wq_l, Wk_l, Wv_l,
          bq + l * Dm, bk + l * Dm, bv + l * Dm,
          qbuf, kcL, vcL, t);

      // attention over cached positions 0..t
      attn_kernel<<<Bsz * Hc, 32, 0, stream>>>(qbuf, kcL, vcL, ctxb, t);

      // output projection + residual -> xbuf
      gemm_out_kernel<<<8, 128, 0, stream>>>(
          ctxb, Wo_l, bo + l * Dm, hbuf, xbuf, (bf16*)nullptr, Dm, Dm, 0);
      layernorm_kernel<<<Bsz, 256, 0, stream>>>(
          xbuf, ln1s + l * Dm, ln1b + l * Dm, hbuf, hb);

      // FFN: relu(h@W1+b1) @ W2 + b2 + residual
      gemm_out_kernel<<<32, 128, 0, stream>>>(
          hb, W1_l, b1 + l * DFF, (const float*)nullptr,
          (float*)nullptr, f1b, Dm, DFF, 1);
      gemm_out_kernel<<<8, 128, 0, stream>>>(
          f1b, W2_l, b2 + l * Dm, hbuf, xbuf, (bf16*)nullptr, DFF, Dm, 0);
      layernorm_kernel<<<Bsz, 256, 0, stream>>>(
          xbuf, ln2s + l * Dm, ln2b + l * Dm, hbuf, hb);
    }

    final_kernel<<<1, 128, 0, stream>>>(hbuf, Wf, bfv, out, t);
  }
}
